// Decomposeable_35459249996696
// MI455X (gfx1250) — matrix operand order repaired and checked by reading, not compiled, not run on hardware
//
#include <hip/hip_runtime.h>

// ---------------- problem constants (match reference) ----------------
#define NB   128      // batch
#define NL   256      // seq len
#define NE   300      // embed
#define NP   200      // proj (== att dim)
#define NC   3        // classes
#define NBL  (NB * NL)

// padded dims (multiples of 32) so GEMMs have NO edge handling
#define NEp  320      // 300 -> 320
#define NPp  224      // 200 -> 224
#define NVp  416      // 400 -> 416

// ---------------- CDNA5 WMMA types ----------------
typedef __attribute__((ext_vector_type(16))) __bf16 v16bf;
typedef __attribute__((ext_vector_type(8)))  __bf16 v8bf;
typedef __attribute__((ext_vector_type(4)))  __bf16 v4bf;
typedef __attribute__((ext_vector_type(8)))  float  v8f;
typedef __attribute__((ext_vector_type(4)))  float  vf4;

// ---------------- GEMM tiling ----------------
#define BM 128         // macro-tile rows (8 waves x 16)
#define BN 32          // macro-tile cols (2 WMMA tiles per wave)
#define BK 32          // K step (one WMMA K)
#define LDS_STRIDE 40  // BK + 8 halves pad -> 80B row stride (16B aligned, bank-spread)

#define SHUF16(lo, hi) __builtin_shufflevector(lo, hi, 0, 1, 2, 3, 4, 5, 6, 7, \
                                               8, 9, 10, 11, 12, 13, 14, 15)

// -------------------------------------------------------------------
// Branch-free batched GEMM (all dims are exact tile multiples):
//   C[b] = op(A[b] x B[b] (+C[b] if ACCUM) + bias)
// f32 staged through LDS as bf16; f32 accumulate; 2 WMMAs/wave/step.
// Global loads for step k+1 are issued before the WMMAs of step k.
// TRANSB/ACCUM are compile-time so the hot loop has no runtime branch.
// -------------------------------------------------------------------
template <int TRANSB, int ACCUM>
__global__ __launch_bounds__(256) void gemm_bf16_wmma(
    const float* __restrict__ A, const float* __restrict__ B,
    const float* __restrict__ bias, float* __restrict__ C,
    int M, int N, int K,
    int lda, int ldb, int ldc,
    long long sA, long long sB, long long sC,
    int relu) {
  __shared__ __bf16 As[BM][LDS_STRIDE];
  __shared__ __bf16 Bs[BN][LDS_STRIDE];

  const float* Ab = A + (long long)blockIdx.z * sA;
  const float* Bb = B + (long long)blockIdx.z * sB;
  float*       Cb = C + (long long)blockIdx.z * sC;

  int tid  = threadIdx.x;
  int lane = tid & 31;
  int wave = tid >> 5;          // 0..7 -> 16-row strip
  int brow = blockIdx.y * BM;
  int bcol = blockIdx.x * BN;

  // ---- staging plans (coalesced, aligned float4) ----
  int ar = tid >> 1;            // 0..127 : A row
  int ac = (tid & 1) * 16;      // 0 / 16 : A col base (16 floats)
  const float* aptr = Ab + (long long)(brow + ar) * lda + ac;

  // B staging: NN -> (k = tid>>3 row, 4 n's), NT -> (n = tid>>3 row, 4 k's)
  int brow_s = tid >> 3;        // 0..31
  int boff_s = (tid & 7) * 4;   // 0..28
  const float* bptr = TRANSB
      ? Bb + (long long)(bcol + brow_s) * ldb + boff_s
      : Bb + (long long)brow_s * ldb + (bcol + boff_s);

  // ---- per-lane fragment addresses (CDNA5 WMMA VGPR layouts) ----
  int fa_r = wave * 16 + (lane & 15);
  int fa_k = (lane & 16) ? 16 : 0;    // halves
  int fb_k = (lane & 16) ? 16 : 0;    // halves
  int fn0  = lane & 15;

  // ---- per-lane C coords ----
  int cm  = brow + wave * 16 + ((lane & 16) ? 8 : 0);
  int cn0 = bcol + (lane & 15);
  int cn1 = cn0 + 16;

  v8f acc0 = {}, acc1 = {};
  if (ACCUM) {
#pragma unroll
    for (int i = 0; i < 8; ++i) {
      acc0[i] = Cb[(long long)(cm + i) * ldc + cn0];
      acc1[i] = Cb[(long long)(cm + i) * ldc + cn1];
    }
  }

  // ---- register prefetch (k0 = 0) ----
  vf4 arg0, arg1, arg2, arg3, brg;
  arg0 = *(const vf4*)(aptr + 0);
  arg1 = *(const vf4*)(aptr + 4);
  arg2 = *(const vf4*)(aptr + 8);
  arg3 = *(const vf4*)(aptr + 12);
  brg  = *(const vf4*)bptr;

  for (int k0 = 0; k0 < K; k0 += BK) {
    // ---- convert staged regs -> bf16 LDS ----
    {
      v8bf t0, t1;
#pragma unroll
      for (int i = 0; i < 4; ++i) { t0[i] = (__bf16)arg0[i]; t0[4 + i] = (__bf16)arg1[i]; }
#pragma unroll
      for (int i = 0; i < 4; ++i) { t1[i] = (__bf16)arg2[i]; t1[4 + i] = (__bf16)arg3[i]; }
      *(v8bf*)&As[ar][ac]     = t0;
      *(v8bf*)&As[ar][ac + 8] = t1;
      if (TRANSB) {
        v4bf t2;
#pragma unroll
        for (int i = 0; i < 4; ++i) t2[i] = (__bf16)brg[i];
        *(v4bf*)&Bs[brow_s][boff_s] = t2;      // Bs[n][k], contiguous k
      } else {
#pragma unroll
        for (int i = 0; i < 4; ++i) Bs[boff_s + i][brow_s] = (__bf16)brg[i];  // transpose
      }
    }
    __syncthreads();

    // ---- issue next tile's global loads (hidden behind WMMAs) ----
    if (k0 + BK < K) {
      const float* p = aptr + (k0 + BK);
      arg0 = *(const vf4*)(p + 0);
      arg1 = *(const vf4*)(p + 4);
      arg2 = *(const vf4*)(p + 8);
      arg3 = *(const vf4*)(p + 12);
      brg  = TRANSB ? *(const vf4*)(bptr + (k0 + BK))
                    : *(const vf4*)(bptr + (long long)(k0 + BK) * ldb);
    }

    // ---- fragments + 2 WMMAs (A fragment reused) ----
    v8bf a_lo = *(const v8bf*)&As[fa_r][fa_k];
    v8bf a_hi = *(const v8bf*)&As[fa_r][fa_k + 8];
    v16bf af  = SHUF16(a_lo, a_hi);
    v8bf b0l = *(const v8bf*)&Bs[fn0][fb_k];
    v8bf b0h = *(const v8bf*)&Bs[fn0][fb_k + 8];
    v16bf bf0 = SHUF16(b0l, b0h);
    v8bf b1l = *(const v8bf*)&Bs[fn0 + 16][fb_k];
    v8bf b1h = *(const v8bf*)&Bs[fn0 + 16][fb_k + 8];
    v16bf bf1 = SHUF16(b1l, b1h);

    acc0 = __builtin_amdgcn_wmma_f32_16x16x32_bf16(false, af, false, bf0,
                                                   (short)0, acc0, false, false);
    acc1 = __builtin_amdgcn_wmma_f32_16x16x32_bf16(false, af, false, bf1,
                                                   (short)0, acc1, false, false);
    __syncthreads();
  }

  // ---- epilogue (no bounds: everything padded) ----
#pragma unroll
  for (int i = 0; i < 8; ++i) {
    float v0 = acc0[i], v1 = acc1[i];
    if (bias) { v0 += bias[cn0]; v1 += bias[cn1]; }
    if (relu) { v0 = fmaxf(v0, 0.0f); v1 = fmaxf(v1, 0.0f); }
    Cb[(long long)(cm + i) * ldc + cn0] = v0;
    Cb[(long long)(cm + i) * ldc + cn1] = v1;
  }
}

// ---------------- zero-padded weight / bias copies ----------------
__global__ void pad_matrix_ld(const float* __restrict__ src, float* __restrict__ dst,
                              int rows, int cols, int srcLd, int drows, int dcols) {
  int c = blockIdx.x * blockDim.x + threadIdx.x;
  int r = blockIdx.y;
  if (c >= dcols || r >= drows) return;
  dst[(long long)r * dcols + c] =
      (r < rows && c < cols) ? src[(long long)r * srcLd + c] : 0.0f;
}
__global__ void pad_vec(const float* __restrict__ src, float* __restrict__ dst, int n, int dn) {
  int i = blockIdx.x * blockDim.x + threadIdx.x;
  if (i < dn) dst[i] = (i < n) ? src[i] : 0.0f;
}

// ---------------- sequence mask: mask[b,l] = l < count_nonzero(x[b,:]) ---------
__global__ void seq_mask_kernel(const int* __restrict__ x, float* __restrict__ mask, int L) {
  int b = blockIdx.x, t = threadIdx.x;
  __shared__ int red[NL];
  red[t] = (x[b * L + t] != 0) ? 1 : 0;
  __syncthreads();
  for (int s = L >> 1; s > 0; s >>= 1) {
    if (t < s) red[t] += red[t + s];
    __syncthreads();
  }
  int cnt = red[0];
  mask[b * L + t] = (t < cnt) ? 1.0f : 0.0f;
}

// -------- embedding gather + L2 row normalize, writes padded rows --------
__global__ void embed_gather_norm(const int* __restrict__ x, const float* __restrict__ emb,
                                  float* __restrict__ e, int D, int Dp) {
  int tok = blockIdx.x;
  int idx = x[tok];
  const float* row = emb + (long long)idx * D;
  __shared__ float red[128];
  float s = 0.0f;
  for (int d = threadIdx.x; d < D; d += blockDim.x) {
    float v = row[d];
    s += v * v;
  }
  red[threadIdx.x] = s;
  __syncthreads();
  for (int st = 64; st > 0; st >>= 1) {
    if (threadIdx.x < st) red[threadIdx.x] += red[threadIdx.x + st];
    __syncthreads();
  }
  float inv = rsqrtf(red[0] + 1e-30f);
  for (int d = threadIdx.x; d < Dp; d += blockDim.x)
    e[(long long)tok * Dp + d] = (d < D) ? row[d] * inv : 0.0f;
}

// ---------------- row softmax (axis=2), optional dist-bias or mult-masks -------
__global__ void softmax_rows(const float* __restrict__ in, float* __restrict__ out,
                             const float* __restrict__ mrow, const float* __restrict__ mcol,
                             const float* __restrict__ bdist, int L) {
  int i = blockIdx.x, b = blockIdx.y, j = threadIdx.x;
  __shared__ float red[NL];
  long long base = ((long long)b * L + i) * L;
  float v = in[base + j];
  if (mrow) v *= mrow[b * L + i] * mcol[b * L + j];
  if (bdist) {
    int d = i - j; if (d < 0) d = -d;
    if (d >= 10) v += bdist[0];
  }
  red[j] = v; __syncthreads();
  for (int s = L >> 1; s > 0; s >>= 1) {
    if (j < s) red[j] = fmaxf(red[j], red[j + s]);
    __syncthreads();
  }
  float mx = red[0]; __syncthreads();
  float e = __expf(v - mx);
  red[j] = e; __syncthreads();
  for (int s = L >> 1; s > 0; s >>= 1) {
    if (j < s) red[j] += red[j + s];
    __syncthreads();
  }
  out[base + j] = e / red[0];
}

// ------- column softmax (axis=1) written TRANSPOSED: out[b,i,k] = sm_k(in[b,k,i]*m) ----
__global__ void softmax_colsT(const float* __restrict__ in, float* __restrict__ out,
                              const float* __restrict__ m1, const float* __restrict__ m2, int L) {
  int i = blockIdx.x, b = blockIdx.y, k = threadIdx.x;
  __shared__ float red[NL];
  float v = in[((long long)b * L + k) * L + i];
  v *= m1[b * L + k] * m2[b * L + i];
  red[k] = v; __syncthreads();
  for (int s = L >> 1; s > 0; s >>= 1) {
    if (k < s) red[k] = fmaxf(red[k], red[k + s]);
    __syncthreads();
  }
  float mx = red[0]; __syncthreads();
  float e = __expf(v - mx);
  red[k] = e; __syncthreads();
  for (int s = L >> 1; s > 0; s >>= 1) {
    if (k < s) red[k] += red[k + s];
    __syncthreads();
  }
  out[((long long)b * L + i) * L + k] = e / red[0];
}

// ---------------- masked sum pooling over L ------------------------------------
__global__ void pool_kernel(const float* __restrict__ v, const float* __restrict__ mask,
                            float* __restrict__ pooled, int L, int D, int outStride, int outOff) {
  int d = blockIdx.x * blockDim.x + threadIdx.x;
  int b = blockIdx.y;
  if (d >= D) return;
  float s = 0.0f;
  for (int l = 0; l < L; ++l)
    s += v[((long long)b * L + l) * D + d] * mask[b * L + l];
  pooled[(long long)b * outStride + outOff + d] = s;
}

// -------- final classifier: concat([v1,v2]) @ Wg + bg (skips pad columns) ------
__global__ void final_kernel(const float* __restrict__ pooled, const float* __restrict__ Wg,
                             const float* __restrict__ bg, float* __restrict__ out) {
  int t = blockIdx.x * blockDim.x + threadIdx.x;
  if (t >= NB * NC) return;
  int b = t / NC, c = t % NC;
  const float* pr = pooled + (long long)b * (2 * NVp);
  float s = bg[c];
  for (int d = 0; d < 2 * NP; ++d) s += pr[d] * Wg[d * NC + c];
  for (int d = 0; d < 2 * NP; ++d) s += pr[NVp + d] * Wg[(2 * NP + d) * NC + c];
  out[b * NC + c] = s;
}

// -------------------------------------------------------------------
static void gemm(hipStream_t stream, const float* A, const float* B, const float* bias,
                 float* C, int M, int N, int K, int lda, int ldb, int ldc,
                 long long sA, long long sB, long long sC, int batch,
                 int transB, int accum, int relu) {
  dim3 grid(N / BN, M / BM, batch);
  if (transB) {
    if (accum) gemm_bf16_wmma<1, 1><<<grid, 256, 0, stream>>>(A, B, bias, C, M, N, K, lda, ldb, ldc, sA, sB, sC, relu);
    else       gemm_bf16_wmma<1, 0><<<grid, 256, 0, stream>>>(A, B, bias, C, M, N, K, lda, ldb, ldc, sA, sB, sC, relu);
  } else {
    if (accum) gemm_bf16_wmma<0, 1><<<grid, 256, 0, stream>>>(A, B, bias, C, M, N, K, lda, ldb, ldc, sA, sB, sC, relu);
    else       gemm_bf16_wmma<0, 0><<<grid, 256, 0, stream>>>(A, B, bias, C, M, N, K, lda, ldb, ldc, sA, sB, sC, relu);
  }
}

extern "C" void kernel_launch(void* const* d_in, const int* in_sizes, int n_in,
                              void* d_out, int out_size, void* d_ws, size_t ws_size,
                              hipStream_t stream) {
  const int*   x1    = (const int*)d_in[0];
  const int*   x2    = (const int*)d_in[1];
  const float* emb   = (const float*)d_in[2];
  const float* Wi    = (const float*)d_in[3];
  const float* bi    = (const float*)d_in[4];
  const float* bdist = (const float*)d_in[5];
  const float* Wp    = (const float*)d_in[6];
  const float* bp    = (const float*)d_in[7];
  const float* Wa    = (const float*)d_in[8];
  const float* ba    = (const float*)d_in[9];
  const float* Wc    = (const float*)d_in[10];
  const float* bc    = (const float*)d_in[11];
  const float* Wg    = (const float*)d_in[12];
  const float* bg    = (const float*)d_in[13];
  float* out = (float*)d_out;
  float* ws  = (float*)d_ws;

  // ---- workspace layout (floats), padded leading dims ----
  float* e      = ws;                               // NBL*NEp
  float* f      = e      + (size_t)NBL * NEp;       // NBL*NPp (intra FC / later alpha)
  float* att    = f      + (size_t)NBL * NPp;       // NB*NL*NL (intra att / cross sim)
  float* xp     = att    + (size_t)NB * NL * NL;    // NBL*NEp (intra aligned / later beta)
  float* p1     = xp     + (size_t)NBL * NEp;       // NBL*NPp
  float* p2     = p1     + (size_t)NBL * NPp;       // NBL*NPp
  float* a1     = p2     + (size_t)NBL * NPp;       // NBL*NPp
  float* a2     = a1     + (size_t)NBL * NPp;       // NBL*NPp
  float* SM     = a2     + (size_t)NBL * NPp;       // NB*NL*NL (softmaxed sim)
  float* vbuf   = SM     + (size_t)NB * NL * NL;    // NBL*NVp (compare output, reused)
  float* m1     = vbuf   + (size_t)NBL * NVp;       // NBL
  float* m2     = m1     + (size_t)NBL;             // NBL
  float* pooled = m2     + (size_t)NBL;             // NB * 2*NVp
  float* Wip    = pooled + (size_t)NB * 2 * NVp;    // NEp*NPp
  float* WpT    = Wip    + (size_t)NEp * NPp;       // NEp*NPp (rows 0..299 of Wp)
  float* WpB    = WpT    + (size_t)NEp * NPp;       // NEp*NPp (rows 300..599)
  float* Wap    = WpB    + (size_t)NEp * NPp;       // NPp*NPp
  float* WcT    = Wap    + (size_t)NPp * NPp;       // NPp*NVp (rows 0..199 of Wc)
  float* WcB    = WcT    + (size_t)NPp * NVp;       // NPp*NVp (rows 200..399)
  float* bip    = WcB    + (size_t)NPp * NVp;       // NPp
  float* bpp    = bip    + NPp;                     // NPp
  float* bap    = bpp    + NPp;                     // NPp
  float* bcp    = bap    + NPp;                     // NVp

  // ---- pad weights/biases once per launch (tiny) ----
  pad_matrix_ld<<<dim3((NPp + 255) / 256, NEp), 256, 0, stream>>>(Wi, Wip, NE, NP, NP, NEp, NPp);
  pad_matrix_ld<<<dim3((NPp + 255) / 256, NEp), 256, 0, stream>>>(Wp, WpT, NE, NP, NP, NEp, NPp);
  pad_matrix_ld<<<dim3((NPp + 255) / 256, NEp), 256, 0, stream>>>(Wp + (size_t)NE * NP, WpB, NE, NP, NP, NEp, NPp);
  pad_matrix_ld<<<dim3((NPp + 255) / 256, NPp), 256, 0, stream>>>(Wa, Wap, NP, NP, NP, NPp, NPp);
  pad_matrix_ld<<<dim3((NVp + 255) / 256, NPp), 256, 0, stream>>>(Wc, WcT, NP, 2 * NP, 2 * NP, NPp, NVp);
  pad_matrix_ld<<<dim3((NVp + 255) / 256, NPp), 256, 0, stream>>>(Wc + (size_t)NP * 2 * NP, WcB, NP, 2 * NP, 2 * NP, NPp, NVp);
  pad_vec<<<1, NPp, 0, stream>>>(bi, bip, NP, NPp);
  pad_vec<<<1, NPp, 0, stream>>>(bp, bpp, NP, NPp);
  pad_vec<<<1, NPp, 0, stream>>>(ba, bap, NP, NPp);
  pad_vec<<<1, NVp, 0, stream>>>(bc, bcp, 2 * NP, NVp);

  // ================= per-side: embed -> intra -> projection =================
  for (int s = 0; s < 2; ++s) {
    const int* xs = s ? x2 : x1;
    float*     ms = s ? m2 : m1;
    float*     ps = s ? p2 : p1;

    seq_mask_kernel<<<NB, NL, 0, stream>>>(xs, ms, NL);
    embed_gather_norm<<<NBL, 128, 0, stream>>>(xs, emb, e, NE, NEp);

    // f = relu(e @ Wip + bip)         [NBL,NEp] x [NEp,NPp]
    gemm(stream, e, Wip, bip, f, NBL, NPp, NEp, NEp, NPp, NPp, 0, 0, 0, 1, 0, 0, 1);

    // att = f @ f^T (batched NT)      pad cols are zero -> K=NPp safe
    gemm(stream, f, f, nullptr, att, NL, NL, NPp, NPp, NPp, NL,
         (long long)NL * NPp, (long long)NL * NPp, (long long)NL * NL, NB, 1, 0, 0);

    // softmax(att + dist_bias, axis=2) in place
    softmax_rows<<<dim3(NL, NB), NL, 0, stream>>>(att, att, nullptr, nullptr, bdist, NL);

    // xp = att @ e (batched NN)
    gemm(stream, att, e, nullptr, xp, NL, NEp, NL, NL, NEp, NEp,
         (long long)NL * NL, (long long)NL * NEp, (long long)NL * NEp, NB, 0, 0, 0);

    // p = [e, xp] @ Wp + bp  (split accumulate)
    gemm(stream, e, WpT, nullptr, ps, NBL, NPp, NEp, NEp, NPp, NPp, 0, 0, 0, 1, 0, 0, 0);
    gemm(stream, xp, WpB, bpp, ps, NBL, NPp, NEp, NEp, NPp, NPp, 0, 0, 0, 1, 0, 1, 0);
  }

  // ================= attent: cross similarity + alignments =================
  gemm(stream, p1, Wap, bap, a1, NBL, NPp, NPp, NPp, NPp, NPp, 0, 0, 0, 1, 0, 0, 1);
  gemm(stream, p2, Wap, bap, a2, NBL, NPp, NPp, NPp, NPp, NPp, 0, 0, 0, 1, 0, 0, 1);

  // sim = a1 @ a2^T (batched NT)
  gemm(stream, a1, a2, nullptr, att, NL, NL, NPp, NPp, NPp, NL,
       (long long)NL * NPp, (long long)NL * NPp, (long long)NL * NL, NB, 1, 0, 0);

  // beta = softmax(masked sim, axis=2) @ p2
  softmax_rows<<<dim3(NL, NB), NL, 0, stream>>>(att, SM, m1, m2, nullptr, NL);
  gemm(stream, SM, p2, nullptr, xp /*beta*/, NL, NPp, NL, NL, NPp, NPp,
       (long long)NL * NL, (long long)NL * NPp, (long long)NL * NPp, NB, 0, 0, 0);

  // alpha = softmax(masked sim, axis=1)^T @ p1 (col-softmax written transposed)
  softmax_colsT<<<dim3(NL, NB), NL, 0, stream>>>(att, SM, m1, m2, NL);
  gemm(stream, SM, p1, nullptr, f /*alpha*/, NL, NPp, NL, NL, NPp, NPp,
       (long long)NL * NL, (long long)NL * NPp, (long long)NL * NPp, NB, 0, 0, 0);

  // ================= compare + masked pooling =================
  // v1 = relu([p1, beta] @ Wc + bc)
  gemm(stream, p1, WcT, nullptr, vbuf, NBL, NVp, NPp, NPp, NVp, NVp, 0, 0, 0, 1, 0, 0, 0);
  gemm(stream, xp /*beta*/, WcB, bcp, vbuf, NBL, NVp, NPp, NPp, NVp, NVp, 0, 0, 0, 1, 0, 1, 1);
  pool_kernel<<<dim3((NVp + 255) / 256, NB), 256, 0, stream>>>(vbuf, m1, pooled, NL, NVp, 2 * NVp, 0);

  // v2 = relu([p2, alpha] @ Wc + bc)
  gemm(stream, p2, WcT, nullptr, vbuf, NBL, NVp, NPp, NPp, NVp, NVp, 0, 0, 0, 1, 0, 0, 0);
  gemm(stream, f /*alpha*/, WcB, bcp, vbuf, NBL, NVp, NPp, NPp, NVp, NVp, 0, 0, 0, 1, 0, 1, 1);
  pool_kernel<<<dim3((NVp + 255) / 256, NB), 256, 0, stream>>>(vbuf, m2, pooled, NL, NVp, 2 * NVp, NVp);

  // ================= aggregate classifier =================
  final_kernel<<<(NB * NC + 127) / 128, 128, 0, stream>>>(pooled, Wg, bg, out);
}
